// RunRingFlashAttn_63041529970731
// MI455X (gfx1250) — matrix operand order repaired and checked by reading, not compiled, not run on hardware
//
#include <hip/hip_runtime.h>

// CDNA5 / gfx1250 flash-attention forward.
// [B,S,H,D] = [2,2048,16,64], fp32 in/out; bf16 WMMA 16x16x32, fp32 accum.
// S^T = K*Q^T and O^T = V^T*P^T formulation (softmax stats one-per-lane).
// kv chunk = 64, double-buffered LDS staging, software-pipelined loads.
// Row-sum of P computed on the matrix pipe via an all-ones A operand.

typedef __attribute__((ext_vector_type(16))) __bf16 v16bf;
typedef __attribute__((ext_vector_type(8)))  __bf16 v8bf;
typedef __attribute__((ext_vector_type(2)))  __bf16 v2bf;
typedef __attribute__((ext_vector_type(8)))  float  v8f;
typedef __attribute__((ext_vector_type(4)))  float  v4f;
typedef __attribute__((ext_vector_type(8)))  int    v8i;

namespace {
constexpr int kB = 2;
constexpr int kS = 2048;
constexpr int kH = 16;
constexpr int kD = 64;
constexpr int kKv = 64;                      // kv positions per chunk
constexpr int kNChunks = kS / kKv;           // 32
// fold softmax scale AND log2(e) into Q so scores live in exp2 domain
constexpr float kQScale = 0.125f * 1.44269504088896340736f;
}

static __device__ __forceinline__ int pack_bf16(float a, float b) {
    v2bf t; t[0] = (__bf16)a; t[1] = (__bf16)b;
    return __builtin_bit_cast(int, t);
}

// Assemble a WMMA 16-bit A operand from two contiguous 8-element LDS runs.
static __device__ __forceinline__ v16bf load_a(const __bf16* base) {
    v8bf r1 = *(const v8bf*)base;
    v8bf r2 = *(const v8bf*)(base + 8);
    v16bf a;
    #pragma unroll
    for (int i = 0; i < 8; ++i) { a[i] = r1[i]; a[i + 8] = r2[i]; }
    return a;
}

// Build B = P^T operand (lane: q-col; elem e -> k = hi*16+e) from two
// C-layout score tiles; bf16-pack first so only 4 dwords/tile cross lanes.
static __device__ __forceinline__ v16bf build_p(const v8f& s0, const v8f& s1,
                                                bool low) {
    int d0[4], d1[4];
    #pragma unroll
    for (int i = 0; i < 4; ++i) {
        d0[i] = pack_bf16(s0[2 * i], s0[2 * i + 1]);
        d1[i] = pack_bf16(s1[2 * i], s1[2 * i + 1]);
    }
    v8i w;
    #pragma unroll
    for (int i = 0; i < 4; ++i) {
        int p0 = __shfl_xor(d0[i], 16, 32);
        int p1 = __shfl_xor(d1[i], 16, 32);
        w[i]     = low ? d0[i] : p1;
        w[i + 4] = low ? p0 : d1[i];
    }
    return __builtin_bit_cast(v16bf, w);
}

__global__ __launch_bounds__(256)
void fa_fwd_wmma_kernel(const float* __restrict__ q,
                        const float* __restrict__ k,
                        const float* __restrict__ v,
                        float* __restrict__ out)
{
    // Double-buffered: K row-major [k][d], V transposed [d][k]; 32 KB total.
    __shared__ __align__(16) __bf16 ldsK[2][kKv * kD];
    __shared__ __align__(16) __bf16 ldsVT[2][kD * kKv];

    const int tid  = threadIdx.x;
    const int lane = tid & 31;
    const int wave = tid >> 5;
    const int lo16 = lane & 15;
    const int hi   = lane >> 4;
    const bool low = (hi == 0);

    // block -> (b, h, 128-row q block); grid = 2*16*16 = 512
    const int bid  = blockIdx.x;
    const int b    = bid >> 8;
    const int h    = (bid >> 4) & 15;
    const int qblk = bid & 15;
    const int qrow = qblk * 128 + wave * 16 + lo16;

    // ---- register-resident Q^T B-operands, pre-scaled ----
    v16bf bQ[2];
    {
        const float* qp = q + ((size_t)(b * kS + qrow) * kH + h) * kD;
        #pragma unroll
        for (int c = 0; c < 2; ++c) {
            const float* p = qp + c * 32 + hi * 16;
            #pragma unroll
            for (int i = 0; i < 4; ++i) {
                v4f f = *(const v4f*)(p + i * 4);
                #pragma unroll
                for (int j = 0; j < 4; ++j)
                    bQ[c][i * 4 + j] = (__bf16)(f[j] * kQScale);
            }
        }
    }

    // Constant all-ones A operand: sums P^T columns on the matrix pipe.
    v16bf aOnes;
    #pragma unroll
    for (int i = 0; i < 16; ++i) aOnes[i] = (__bf16)1.0f;

    v8f accO[4];
    #pragma unroll
    for (int dt = 0; dt < 4; ++dt) {
        v8f z = {0.f, 0.f, 0.f, 0.f, 0.f, 0.f, 0.f, 0.f};
        accO[dt] = z;
    }
    float m    = -3.0e38f;   // running max, exp2 domain
    float lsum = 0.0f;

    // ---- staging assignments ----
    // K: thread covers row tid>>2 (64 rows, 4 threads/row), 16 d values.
    const int krowK = tid >> 2;
    const int ksegK = (tid & 3) * 16;
    const size_t kBase = ((size_t)(b * kS + krowK) * kH + h) * kD + ksegK;
    // V: thread covers k-pair (2*kp, 2*kp+1), 8 d values -> b32 LDS stores.
    const int vkp = tid & 31;
    const int vds = (tid >> 5) * 8;
    const size_t rowStride = (size_t)kH * kD;
    const size_t vBase = ((size_t)(b * kS + 2 * vkp) * kH + h) * kD + vds;
    const size_t chunkStride = (size_t)kKv * kH * kD;

    v4f kr[4], va[2], vb_[2];

    auto gload = [&](int chunk) {
        const float* kp = k + kBase + (size_t)chunk * chunkStride;
        #pragma unroll
        for (int i = 0; i < 4; ++i) kr[i] = *(const v4f*)(kp + i * 4);
        const float* vp = v + vBase + (size_t)chunk * chunkStride;
        va[0]  = *(const v4f*)vp;
        va[1]  = *(const v4f*)(vp + 4);
        vb_[0] = *(const v4f*)(vp + rowStride);
        vb_[1] = *(const v4f*)(vp + rowStride + 4);
    };

    auto cstore = [&](int buf) {
        v8bf s;
        #pragma unroll
        for (int j = 0; j < 4; ++j) { s[j] = (__bf16)kr[0][j]; s[4 + j] = (__bf16)kr[1][j]; }
        *(v8bf*)&ldsK[buf][krowK * kD + ksegK] = s;
        #pragma unroll
        for (int j = 0; j < 4; ++j) { s[j] = (__bf16)kr[2][j]; s[4 + j] = (__bf16)kr[3][j]; }
        *(v8bf*)&ldsK[buf][krowK * kD + ksegK + 8] = s;
        #pragma unroll
        for (int j = 0; j < 8; ++j) {
            const float a  = (j < 4) ? va[0][j]  : va[1][j - 4];
            const float bb = (j < 4) ? vb_[0][j] : vb_[1][j - 4];
            *(int*)&ldsVT[buf][(vds + j) * kKv + 2 * vkp] = pack_bf16(a, bb);
        }
    };

    // ---- pipeline prologue ----
    gload(0);
    cstore(0);
    __syncthreads();

    for (int chunk = 0; chunk < kNChunks; ++chunk) {
        const int cur = chunk & 1;
        if (chunk + 1 < kNChunks) gload(chunk + 1);   // loads fly under compute

        // ---- S^T = K * Q^T : four 16-k tiles ----
        v8f st[4];
        #pragma unroll
        for (int kt = 0; kt < 4; ++kt) {
            v8f acc = {0.f, 0.f, 0.f, 0.f, 0.f, 0.f, 0.f, 0.f};
            #pragma unroll
            for (int c = 0; c < 2; ++c) {
                v16bf a = load_a(&ldsK[cur][(kt * 16 + lo16) * kD + c * 32 + hi * 16]);
                acc = __builtin_amdgcn_wmma_f32_16x16x32_bf16(
                          false, a, false, bQ[c], (short)0, acc, false, false);
            }
            st[kt] = acc;
        }

        // ---- online softmax over 64-k chunk (exp2 domain) ----
        float cmax = m;
        #pragma unroll
        for (int kt = 0; kt < 4; ++kt)
            #pragma unroll
            for (int e = 0; e < 8; ++e)
                cmax = fmaxf(cmax, st[kt][e]);
        cmax = fmaxf(cmax, __shfl_xor(cmax, 16, 32));
        const float alpha = __builtin_amdgcn_exp2f(m - cmax);
        m = cmax;
        #pragma unroll
        for (int kt = 0; kt < 4; ++kt)
            #pragma unroll
            for (int e = 0; e < 8; ++e)
                st[kt][e] = __builtin_amdgcn_exp2f(st[kt][e] - m);

        // ---- rescale previous accumulators BEFORE adding this chunk ----
        lsum *= alpha;
        #pragma unroll
        for (int dt = 0; dt < 4; ++dt)
            #pragma unroll
            for (int e = 0; e < 8; ++e)
                accO[dt][e] *= alpha;

        // ---- P^T operands (k 0..31 and 32..63) ----
        v16bf bpA = build_p(st[0], st[1], low);
        v16bf bpB = build_p(st[2], st[3], low);

        // ---- row sums of P on the matrix pipe (ones-row trick) ----
        v8f accL = {0.f, 0.f, 0.f, 0.f, 0.f, 0.f, 0.f, 0.f};
        accL = __builtin_amdgcn_wmma_f32_16x16x32_bf16(
                   false, aOnes, false, bpA, (short)0, accL, false, false);
        accL = __builtin_amdgcn_wmma_f32_16x16x32_bf16(
                   false, aOnes, false, bpB, (short)0, accL, false, false);

        // ---- O^T += V^T * P^T (also hides the accL->VALU hazard) ----
        #pragma unroll
        for (int dt = 0; dt < 4; ++dt) {
            const __bf16* vrow = &ldsVT[cur][(dt * 16 + lo16) * kKv];
            accO[dt] = __builtin_amdgcn_wmma_f32_16x16x32_bf16(
                           false, load_a(vrow + hi * 16), false, bpA,
                           (short)0, accO[dt], false, false);
            accO[dt] = __builtin_amdgcn_wmma_f32_16x16x32_bf16(
                           false, load_a(vrow + 32 + hi * 16), false, bpB,
                           (short)0, accO[dt], false, false);
        }

        // accL sums over all 64 k (both half-waves) -> no shuffle needed.
        lsum += accL[0];

        if (chunk + 1 < kNChunks) cstore(cur ^ 1);    // fill other buffer
        __syncthreads();
    }

    // ---- epilogue: normalize + contiguous stores ----
    const float inv = 1.0f / lsum;
    float* op = out + ((size_t)(b * kS + qrow) * kH + h) * kD;
    #pragma unroll
    for (int dt = 0; dt < 4; ++dt) {
        const int d0 = dt * 16 + hi * 8;
        v4f o0, o1;
        #pragma unroll
        for (int j = 0; j < 4; ++j) {
            o0[j] = accO[dt][j] * inv;
            o1[j] = accO[dt][4 + j] * inv;
        }
        *(v4f*)(op + d0)     = o0;
        *(v4f*)(op + d0 + 4) = o1;
    }
}

extern "C" void kernel_launch(void* const* d_in, const int* in_sizes, int n_in,
                              void* d_out, int out_size, void* d_ws, size_t ws_size,
                              hipStream_t stream) {
    const float* q = (const float*)d_in[0];
    const float* k = (const float*)d_in[1];
    const float* v = (const float*)d_in[2];
    float* out = (float*)d_out;

    dim3 grid(kB * kH * (kS / 128));   // 512 workgroups
    dim3 block(256);                   // 8 waves (wave32)
    hipLaunchKernelGGL(fa_fwd_wmma_kernel, grid, block, 0, stream, q, k, v, out);
}
